// LanguageModel_57346403336502
// MI455X (gfx1250) — matrix operand order repaired and checked by reading, not compiled, not run on hardware
//
#include <hip/hip_runtime.h>

#define B_  32
#define S_  512
#define H_  1024
#define G_  4096   // 4*H
#define V_  32000

typedef __attribute__((ext_vector_type(16))) __bf16          v16bf;
typedef __attribute__((ext_vector_type(8)))  __bf16          v8bf;
typedef __attribute__((ext_vector_type(16))) unsigned short  v16us;
typedef __attribute__((ext_vector_type(8)))  float           v8f;

// ---------- helpers ----------

__device__ __forceinline__ unsigned short f2bf_rne(float f) {
  unsigned u = __builtin_bit_cast(unsigned, f);
  u = u + 0x7FFFu + ((u >> 16) & 1u);   // round-to-nearest-even
  return (unsigned short)(u >> 16);
}
__device__ __forceinline__ __bf16 f2bf(float f) {
  return __builtin_bit_cast(__bf16, f2bf_rne(f));
}

// A-matrix 16x32 bf16 fragment in contiguous k order (same as load_b_tile), loaded straight from a bf16 row:
//   lanes 0-15 : K = k0+[0..15]   (row = lane)
//   lanes 16-31: K = k0+[16..31]  (row = lane-16)
// Two contiguous 16B loads + register concat; zero conversion VALU.
__device__ __forceinline__ v16bf load_a_tile_bf(const __bf16* rowbase, int k0, int laneHi) {
  int kb = k0 + (laneHi ? 16 : 0);
  v8bf lo = *(const v8bf*)(rowbase + kb);
  v8bf hi = *(const v8bf*)(rowbase + kb + 8);
  return __builtin_shufflevector(lo, hi, 0, 1, 2, 3, 4, 5, 6, 7,
                                 8, 9, 10, 11, 12, 13, 14, 15);
}

// B-matrix 32x16 bf16 fragment: lane n holds 16 contiguous K values
// (lanes 0-15 -> K = k0..k0+15, lanes 16-31 -> K = k0+16..k0+31).
// WT is pre-transposed [N][K] bf16 -> one contiguous 32B load.
__device__ __forceinline__ v16bf load_b_tile(const __bf16* WT, int n, int k0, int khalf) {
  return *(const v16bf*)(WT + (size_t)n * H_ + k0 + khalf * 16);
}

__device__ __forceinline__ v8f wmma_bf16(v16bf a, v16bf b, v8f c) {
  return __builtin_amdgcn_wmma_f32_16x16x32_bf16(false, a, false, b, (short)0, c, false, false);
}

__device__ __forceinline__ float sigmoidf_(float x) { return 1.f / (1.f + __expf(-x)); }

// ---------- weight cast + transpose:  W[K=1024][N=4096] f32 -> WT[N][K] bf16 ----------

__global__ __launch_bounds__(256) void cast_transpose_kernel(const float* __restrict__ W,
                                                             __bf16* __restrict__ WT) {
  __shared__ float tile[32][33];
  int tx = threadIdx.x & 31, ty = threadIdx.x >> 5;       // 32x8
  int nb = blockIdx.x * 32, kb = blockIdx.y * 32;
#pragma unroll
  for (int i = 0; i < 4; ++i)
    tile[ty + i * 8][tx] = W[(size_t)(kb + ty + i * 8) * G_ + nb + tx];
  __syncthreads();
#pragma unroll
  for (int i = 0; i < 4; ++i)
    WT[(size_t)(nb + ty + i * 8) * H_ + kb + tx] = f2bf(tile[tx][ty + i * 8]);
}

// ---------- embedding pre-gather + cast: xemb[b][t][:] = bf16(embedding[x[b][t]][:]) ----------

__global__ __launch_bounds__(256) void gather_cast_kernel(const int* __restrict__ x,
                                                          const float* __restrict__ emb,
                                                          __bf16* __restrict__ xemb) {
  int bt = blockIdx.x;                       // b*S + t
  const float* src = emb + (size_t)x[bt] * H_;
  __bf16* dst = xemb + (size_t)bt * H_;
  for (int i = threadIdx.x; i < H_; i += 256) dst[i] = f2bf(src[i]);
}

__global__ __launch_bounds__(256) void zero_kernel(float* __restrict__ p, int n) {
  int i = blockIdx.x * 256 + threadIdx.x;
  if (i < n) p[i] = 0.f;
}

// ---------- fused LSTM cell step (one layer, one timestep) ----------
// grid = H/16 blocks (16 hidden units each), block = 128 threads = 4 waves.
// Wave w computes z[:, w*1024 + jblk .. +16)  (gate w of {i,f,g,o}),
// z = a @ Wx + h_prev @ Wh + b, accumulated into one pair of WMMA accumulators.
// All matmul operands are bf16 in memory -> inner loop = b128 loads + v_wmma only.

__global__ __launch_bounds__(128, 1) void lstm_step_kernel(
    const __bf16* __restrict__ a_base,    // layer input rows (bf16)
    long          a_stride,               // row stride in elements
    const __bf16* __restrict__ h_prev,    // [B][H] bf16
    float*        __restrict__ c_state,   // [B][H] f32, in-place
    __bf16*       __restrict__ h_out,     // [B][H] bf16
    const __bf16* __restrict__ WxT,       // [G][H] bf16 (transposed)
    const __bf16* __restrict__ WhT,
    const float*  __restrict__ bias) {
  __shared__ float zl[4][32][16];
  const int tid = threadIdx.x;
  const int wv = tid >> 5, lane = tid & 31;
  const int rlow = lane & 15, laneHi = lane >> 4;
  const int jblk = blockIdx.x * 16;        // hidden-unit base of this block
  const int n0 = wv * H_ + jblk;           // z-column base of this wave's gate

  const __bf16* aRow0 = a_base + (size_t)rlow * a_stride;
  const __bf16* aRow1 = a_base + (size_t)(rlow + 16) * a_stride;
  const __bf16* hRow0 = h_prev + (size_t)rlow * H_;
  const __bf16* hRow1 = h_prev + (size_t)(rlow + 16) * H_;

  v8f acc0 = {}, acc1 = {};
  for (int k0 = 0; k0 < H_; k0 += 32) {
    v16bf bx  = load_b_tile(WxT, n0 + rlow, k0, laneHi);
    v16bf bh  = load_b_tile(WhT, n0 + rlow, k0, laneHi);
    v16bf aa0 = load_a_tile_bf(aRow0, k0, laneHi);
    v16bf ah0 = load_a_tile_bf(hRow0, k0, laneHi);
    v16bf aa1 = load_a_tile_bf(aRow1, k0, laneHi);
    v16bf ah1 = load_a_tile_bf(hRow1, k0, laneHi);
    acc0 = wmma_bf16(aa0, bx, acc0);
    acc0 = wmma_bf16(ah0, bh, acc0);
    acc1 = wmma_bf16(aa1, bx, acc1);
    acc1 = wmma_bf16(ah1, bh, acc1);
  }

  const float bval = bias[n0 + rlow];
  // C/D layout: VGPR r -> row r (lanes 0-15) / row r+8 (lanes 16-31), col = lane&15
#pragma unroll
  for (int r = 0; r < 8; ++r) {
    zl[wv][r + 8 * laneHi][rlow]      = acc0[r] + bval;
    zl[wv][16 + r + 8 * laneHi][rlow] = acc1[r] + bval;
  }
  __syncthreads();

#pragma unroll
  for (int p = tid; p < 512; p += 128) {
    int b = p >> 4, j = p & 15;
    float iv = zl[0][b][j], fv = zl[1][b][j], gv = zl[2][b][j], ov = zl[3][b][j];
    int idx = b * H_ + jblk + j;
    float cn = sigmoidf_(fv) * c_state[idx] + sigmoidf_(iv) * tanhf(gv);
    c_state[idx] = cn;
    h_out[idx]   = f2bf(sigmoidf_(ov) * tanhf(cn));
  }
}

// ---------- final FC: logits = h1 @ Wfc + bfc  (Wfc converted to bf16 on the fly) ----------
// grid = V/128, block = 256 threads = 8 waves, 16 output columns per wave.

__global__ __launch_bounds__(256, 1) void fc_kernel(const __bf16* __restrict__ h,
                                                    const float* __restrict__ Wfc,
                                                    const float* __restrict__ bfc,
                                                    float* __restrict__ logits) {
  const int tid = threadIdx.x, wv = tid >> 5, lane = tid & 31;
  const int rlow = lane & 15, laneHi = lane >> 4;
  const int ncol = blockIdx.x * 128 + wv * 16 + rlow;
  const __bf16* aRow0 = h + (size_t)rlow * H_;
  const __bf16* aRow1 = h + (size_t)(rlow + 16) * H_;

  v8f acc0 = {}, acc1 = {};
  for (int k0 = 0; k0 < H_; k0 += 32) {
    const float* bp = Wfc + (size_t)(k0 + laneHi * 16) * V_ + ncol;
    v16us u;
#pragma unroll
    for (int e = 0; e < 16; ++e) u[e] = f2bf_rne(bp[(size_t)e * V_]);
    v16bf bt = __builtin_bit_cast(v16bf, u);
    v16bf a0 = load_a_tile_bf(aRow0, k0, laneHi);
    v16bf a1 = load_a_tile_bf(aRow1, k0, laneHi);
    acc0 = wmma_bf16(a0, bt, acc0);
    acc1 = wmma_bf16(a1, bt, acc1);
  }
  const float bv = bfc[ncol];
#pragma unroll
  for (int r = 0; r < 8; ++r) {
    logits[(size_t)(r + 8 * laneHi) * V_ + ncol]      = acc0[r] + bv;
    logits[(size_t)(16 + r + 8 * laneHi) * V_ + ncol] = acc1[r] + bv;
  }
}

// ---------- softmax: one block per batch row ----------

__global__ __launch_bounds__(256) void softmax_kernel(const float* __restrict__ logits,
                                                      float* __restrict__ out) {
  __shared__ float red[256];
  const int b = blockIdx.x, tid = threadIdx.x;
  const float* row = logits + (size_t)b * V_;
  float m = -3.4e38f;
  for (int i = tid; i < V_; i += 256) m = fmaxf(m, row[i]);
  red[tid] = m; __syncthreads();
  for (int s = 128; s > 0; s >>= 1) {
    if (tid < s) red[tid] = fmaxf(red[tid], red[tid + s]);
    __syncthreads();
  }
  m = red[0]; __syncthreads();
  float sum = 0.f;
  for (int i = tid; i < V_; i += 256) sum += __expf(row[i] - m);
  red[tid] = sum; __syncthreads();
  for (int s = 128; s > 0; s >>= 1) {
    if (tid < s) red[tid] += red[tid + s];
    __syncthreads();
  }
  const float inv = 1.f / red[0];
  float* orow = out + (size_t)b * V_;
  for (int i = tid; i < V_; i += 256) orow[i] = __expf(row[i] - m) * inv;
}

// ---------- launch ----------

extern "C" void kernel_launch(void* const* d_in, const int* in_sizes, int n_in,
                              void* d_out, int out_size, void* d_ws, size_t ws_size,
                              hipStream_t stream) {
  (void)in_sizes; (void)n_in; (void)out_size; (void)ws_size;
  const int*   x   = (const int*)  d_in[0];
  const float* emb = (const float*)d_in[1];
  const float* Wx0 = (const float*)d_in[2];
  const float* Wh0 = (const float*)d_in[3];
  const float* b0  = (const float*)d_in[4];
  const float* Wx1 = (const float*)d_in[5];
  const float* Wh1 = (const float*)d_in[6];
  const float* b1  = (const float*)d_in[7];
  const float* Wfc = (const float*)d_in[8];
  const float* bfc = (const float*)d_in[9];
  float* out = (float*)d_out;

  char* ws = (char*)d_ws;
  size_t off = 0;
  const size_t wbytes = (size_t)G_ * H_ * sizeof(__bf16);
  __bf16* Wx0T = (__bf16*)(ws + off); off += wbytes;
  __bf16* Wh0T = (__bf16*)(ws + off); off += wbytes;
  __bf16* Wx1T = (__bf16*)(ws + off); off += wbytes;
  __bf16* Wh1T = (__bf16*)(ws + off); off += wbytes;
  __bf16* xemb = (__bf16*)(ws + off); off += (size_t)B_ * S_ * H_ * sizeof(__bf16);
  // zero region starts here: c0, c1 (f32) then h0a, h1a (bf16)
  float*  c0   = (float*)(ws + off);  off += (size_t)B_ * H_ * sizeof(float);
  float*  c1   = (float*)(ws + off);  off += (size_t)B_ * H_ * sizeof(float);
  __bf16* h0a  = (__bf16*)(ws + off); off += (size_t)B_ * H_ * sizeof(__bf16);
  __bf16* h1a  = (__bf16*)(ws + off); off += (size_t)B_ * H_ * sizeof(__bf16);
  __bf16* h0b  = (__bf16*)(ws + off); off += (size_t)B_ * H_ * sizeof(__bf16);
  __bf16* h1b  = (__bf16*)(ws + off); off += (size_t)B_ * H_ * sizeof(__bf16);
  float* logits = (float*)(ws + off); off += (size_t)B_ * V_ * sizeof(float);

  // 1) cast + transpose recurrent weights to bf16 [N][K] (L2-resident: 32 MB)
  dim3 tg(G_ / 32, H_ / 32);
  cast_transpose_kernel<<<tg, 256, 0, stream>>>(Wx0, Wx0T);
  cast_transpose_kernel<<<tg, 256, 0, stream>>>(Wh0, Wh0T);
  cast_transpose_kernel<<<tg, 256, 0, stream>>>(Wx1, Wx1T);
  cast_transpose_kernel<<<tg, 256, 0, stream>>>(Wh1, Wh1T);

  // 2) embedding pre-gather to bf16 (kills all A-side conversion in the scan)
  gather_cast_kernel<<<B_ * S_, 256, 0, stream>>>(x, emb, xemb);

  // 3) zero initial state: c0, c1 (f32) + h0a, h1a (bf16) = 96K dwords
  const int zn = (2 * B_ * H_ * 4 + 2 * B_ * H_ * 2) / 4;
  zero_kernel<<<(zn + 255) / 256, 256, 0, stream>>>(c0, zn);

  // 4) recurrence: 2 fused-cell kernels per timestep, h double-buffered (bf16)
  __bf16* h0buf[2] = {h0a, h0b};
  __bf16* h1buf[2] = {h1a, h1b};
  for (int t = 0; t < S_; ++t) {
    __bf16* h0p = h0buf[t & 1];
    __bf16* h0n = h0buf[(t + 1) & 1];
    __bf16* h1p = h1buf[t & 1];
    __bf16* h1n = h1buf[(t + 1) & 1];
    // layer 0: a = xemb[:, t, :] (row stride S*H), h = h0_prev
    lstm_step_kernel<<<H_ / 16, 128, 0, stream>>>(xemb + (size_t)t * H_, (long)S_ * H_,
                                                  h0p, c0, h0n, Wx0T, Wh0T, b0);
    // layer 1: a = h0_new (row stride H), h = h1_prev
    lstm_step_kernel<<<H_ / 16, 128, 0, stream>>>(h0n, (long)H_,
                                                  h1p, c1, h1n, Wx1T, Wh1T, b1);
  }

  // 5) logits + softmax
  __bf16* h1fin = h1buf[S_ & 1];
  fc_kernel<<<V_ / 128, 256, 0, stream>>>(h1fin, Wfc, bfc, logits);
  softmax_kernel<<<B_, 256, 0, stream>>>(logits, out);
}
